// TransformerEncoder_gos_and_masking_40716289966633
// MI455X (gfx1250) — matrix operand order repaired and checked by reading, not compiled, not run on hardware
//
#include <hip/hip_runtime.h>
#include <hip/hip_bf16.h>
#include <math.h>

// ---------------------------------------------------------------------------
// Problem constants (from reference)
// ---------------------------------------------------------------------------
#define BATCH 8
#define SEQ   1024
#define DMODEL 512
#define NHEAD 8
#define DHEAD 64
#define DMLP  2048
#define ROWS  (BATCH * SEQ)          // 8192
#define QKVN  (3 * NHEAD * DHEAD)    // 1536
#define PRUNE_NUM 64
#define NEGBIG (-1.0e10f)

typedef __bf16 bf16_t;
typedef __attribute__((ext_vector_type(16))) __bf16 v16bf;
typedef __attribute__((ext_vector_type(8)))  float  v8f;

// ---------------------------------------------------------------------------
// Workspace layout (bytes)
// ---------------------------------------------------------------------------
#define OFF_H1     ((size_t)0)          //  8,388,608  bf16 [ROWS, D]
#define OFF_Q      ((size_t)8388608)    //  8,388,608  bf16 [B,H,N,DH] (pre-scaled)
#define OFF_K      ((size_t)16777216)   //  8,388,608  bf16 [B,H,N,DH]
#define OFF_V      ((size_t)25165824)   //  8,388,608  bf16 [B,H] V packed as B-frags
#define OFF_ATTOUT ((size_t)33554432)   //  8,388,608  bf16 [ROWS, H*DH]
#define OFF_X1     ((size_t)41943040)   // 16,777,216  f32  [ROWS, D]
#define OFF_H2     ((size_t)58720256)   //  8,388,608  bf16 [ROWS, D]
#define OFF_FF1    ((size_t)67108864)   // 33,554,432  bf16 [ROWS, MLP]
#define OFF_ROW0   ((size_t)100663296)  //    262,144  f32  [B,H,N] query-0 probs
#define OFF_WQKVP  ((size_t)100925440)  //  1,572,864  bf16 packed Wqkv
#define OFF_WOUTP  ((size_t)102498304)  //    524,288  bf16 packed Wout
#define OFF_W1P    ((size_t)103022592)  //  2,097,152  bf16 packed W1
#define OFF_W2P    ((size_t)105119744)  //  2,097,152  bf16 packed W2

// Output layout (elements, d_out as float*/int* alias)
#define OUT_X_ELEMS   (BATCH * SEQ * DMODEL)   // 4,194,304 f32
#define OUT_PRUNE_OFF OUT_X_ELEMS              // 512 int32
#define OUT_MERGE_OFF (OUT_PRUNE_OFF + BATCH * PRUNE_NUM)
#define OUT_SURV_OFF  (OUT_MERGE_OFF + BATCH)

// ---------------------------------------------------------------------------
// WMMA fragment layouts (CDNA5, wave32)
// A 16x32 (MxK): lanes 0-15 row M=lane, elems 0..15 -> K=k0..k0+15 (order used here)
//                lanes 16-31 same rows, K=k0+16..k0+31.
// B 32x16 (KxN): lanes 0-15 col N=lane&15, elems j -> K=j; lanes 16-31 -> K=16+j
// C/D 16x16 f32: elem r -> row (lane<16 ? r : r+8), col = lane&15
//
// Packed weight layout: tile (nt, kt) of a KxN weight -> 512 bf16:
//   Wp[(nt*KT + kt)*512 + khalf*256 + col*16 + kk], K = kt*32 + khalf*16 + kk
// so each lane's fragment = 32 contiguous bytes (two b128 loads).
// ---------------------------------------------------------------------------
__device__ __forceinline__ v16bf load_a_frag_bf16(const bf16_t* __restrict__ A, int lda,
                                                  int row0, int k0, int lane) {
  const int r  = row0 + (lane & 15);
  const int kb = k0 + ((lane & 16) ? 16 : 0);
  const bf16_t* p = A + (size_t)r * lda + kb;
  v16bf a;
#pragma unroll
  for (int j = 0; j < 8; ++j) {
    a[j]     = p[j];
    a[8 + j] = p[8 + j];
  }
  return a;
}

__device__ __forceinline__ v16bf load_b_frag_packed(const bf16_t* __restrict__ Wp,
                                                    int ktiles, int nt, int kt, int lane) {
  const bf16_t* p = Wp + ((size_t)(nt * ktiles + kt) << 9) +
                    ((lane & 16) ? 256 : 0) + ((lane & 15) << 4);
  v16bf b;
#pragma unroll
  for (int j = 0; j < 16; ++j) b[j] = p[j];
  return b;
}

// Wave-level 32(M) x 64(N) tile: 8 accumulators; each B fragment feeds 2 WMMAs.
__device__ __forceinline__ void wave_gemm_32x64(const bf16_t* __restrict__ A, int lda,
                                                const bf16_t* __restrict__ Wp, int K,
                                                int row0, int col0, int lane, v8f acc[8]) {
  const int ktiles = K >> 5;
  const int nt0 = col0 >> 4;
  for (int kt = 0; kt < ktiles; ++kt) {
    const int k0 = kt << 5;
    v16bf a0 = load_a_frag_bf16(A, lda, row0,      k0, lane);
    v16bf a1 = load_a_frag_bf16(A, lda, row0 + 16, k0, lane);
#pragma unroll
    for (int t = 0; t < 4; ++t) {
      v16bf b = load_b_frag_packed(Wp, ktiles, nt0 + t, kt, lane);
      acc[t]     = __builtin_amdgcn_wmma_f32_16x16x32_bf16(false, a0, false, b,
                                                           (short)0, acc[t], false, false);
      acc[4 + t] = __builtin_amdgcn_wmma_f32_16x16x32_bf16(false, a1, false, b,
                                                           (short)0, acc[4 + t], false, false);
    }
  }
}

// ---------------------------------------------------------------------------
// Weight packing: f32 [K][N] row-major -> bf16 B-fragment tiles (coalesced writes)
// ---------------------------------------------------------------------------
__global__ void k_pack_w(const float* __restrict__ W, bf16_t* __restrict__ Wp,
                         int K, int N) {
  const int idx = blockIdx.x * 256 + threadIdx.x;
  if (idx >= K * N) return;
  const int tile  = idx >> 9;
  const int rem   = idx & 511;
  const int khalf = rem >> 8;
  const int col   = (rem >> 4) & 15;
  const int kk    = rem & 15;
  const int KT    = K >> 5;
  const int nt    = tile / KT;
  const int kt    = tile - nt * KT;
  const int kglob = (kt << 5) + (khalf << 4) + kk;
  const int nglob = (nt << 4) + col;
  Wp[idx] = (__bf16)W[(size_t)kglob * N + nglob];
}

// ---------------------------------------------------------------------------
// LayerNorm: one wave per row (wave32), 16 elems/lane, shuffle reductions
// ---------------------------------------------------------------------------
__global__ void k_layernorm(const float* __restrict__ x, const float* __restrict__ g,
                            const float* __restrict__ bvec, bf16_t* __restrict__ out) {
  const int lane = threadIdx.x & 31;
  const int wave = threadIdx.x >> 5;
  const int row  = blockIdx.x * 8 + wave;
  const float* xr = x + (size_t)row * DMODEL;

  float vals[16];
  float s = 0.f, s2 = 0.f;
#pragma unroll
  for (int j = 0; j < 16; ++j) {
    float t = xr[lane + 32 * j];
    vals[j] = t; s += t; s2 += t * t;
  }
#pragma unroll
  for (int off = 16; off; off >>= 1) {
    s  += __shfl_xor(s, off, 32);
    s2 += __shfl_xor(s2, off, 32);
  }
  const float mu   = s * (1.0f / DMODEL);
  const float var  = s2 * (1.0f / DMODEL) - mu * mu;
  const float rstd = rsqrtf(var + 1e-5f);
#pragma unroll
  for (int j = 0; j < 16; ++j) {
    const int c = lane + 32 * j;
    out[(size_t)row * DMODEL + c] = (__bf16)((vals[j] - mu) * rstd * g[c] + bvec[c]);
  }
}

// ---------------------------------------------------------------------------
// QKV projection: h1(8192x512) @ WqkvP -> q/k [B,H,N,DH] bf16, V packed as B-frags.
// q gets the 1/sqrt(DH) scale folded in.
// ---------------------------------------------------------------------------
__global__ void k_gemm_qkv(const bf16_t* __restrict__ h1, const bf16_t* __restrict__ WqkvP,
                           bf16_t* __restrict__ q, bf16_t* __restrict__ k,
                           bf16_t* __restrict__ vpk) {
  const int lane = threadIdx.x & 31, wave = threadIdx.x >> 5;
  const int row0 = blockIdx.y * 256 + wave * 32;
  const int col0 = blockIdx.x * 64;
  v8f acc[8] = {};
  wave_gemm_32x64(h1, DMODEL, WqkvP, DMODEL, row0, col0, lane, acc);
#pragma unroll
  for (int g = 0; g < 2; ++g) {
#pragma unroll
    for (int t = 0; t < 4; ++t) {
      const int col = col0 + 16 * t + (lane & 15);
      const int which = col >> 9;            // 0=q 1=k 2=v
      const int rem = col & 511;
      const int hh = rem >> 6, dh = rem & 63;
      const float scale = (which == 0) ? 0.125f : 1.0f;  // DH^-0.5
#pragma unroll
      for (int r = 0; r < 8; ++r) {
        const int m  = row0 + g * 16 + r + ((lane & 16) ? 8 : 0);
        const int bb = m >> 10, nn = m & 1023;
        const float val = acc[g * 4 + t][r] * scale;
        if (which == 2) {
          // V packed: [(b*H+h)*(SEQ/32)+c][khalf][dh][kk]
          const int c32 = nn >> 5, khalf = (nn >> 4) & 1, kk = nn & 15;
          vpk[((((size_t)bb * NHEAD + hh) * (SEQ / 32) + c32) * 2 + khalf) * (DHEAD * 16) +
              dh * 16 + kk] = (__bf16)val;
        } else {
          bf16_t* dst = (which == 0) ? q : k;
          dst[((((size_t)bb * NHEAD + hh) * SEQ) + nn) * DHEAD + dh] = (__bf16)val;
        }
      }
    }
  }
}

// ---------------------------------------------------------------------------
// Attention: one wave per (b, h, 16-query tile). fp32 scores (64KB LDS) +
// bf16 probs (32KB LDS); WMMA for both QK^T and P@V.
// ---------------------------------------------------------------------------
__global__ void k_attention(const bf16_t* __restrict__ q, const bf16_t* __restrict__ k,
                            const bf16_t* __restrict__ vpk, const float* __restrict__ mask,
                            bf16_t* __restrict__ attout, float* __restrict__ row0probs) {
  extern __shared__ char smem[];
  float*  s  = (float*)smem;                 // [16][1024] fp32 scores
  bf16_t* pb = (bf16_t*)(smem + 65536);      // [16][1024] bf16 probs
  const int qt = blockIdx.x, h = blockIdx.y, b = blockIdx.z;
  const int lane = threadIdx.x;
  const size_t headbase = ((size_t)b * NHEAD + h) * SEQ;

  // Pre-load the two q A-fragments (K = DH = 64 -> k0 = 0, 32), reused 64x.
  const bf16_t* qb = q + (headbase + qt * 16) * DHEAD;
  v16bf aq[2];
#pragma unroll
  for (int c = 0; c < 2; ++c) aq[c] = load_a_frag_bf16(qb, DHEAD, 0, 32 * c, lane);

  // ---- scores: S[16][1024] ----
  for (int kt = 0; kt < SEQ / 16; ++kt) {
    v8f sc = {};
#pragma unroll
    for (int c = 0; c < 2; ++c) {
      // B 32x16: col = key, row = d (contiguous along DH)
      const int key = kt * 16 + (lane & 15);
      const bf16_t* kp = k + (headbase + key) * DHEAD + 32 * c + ((lane & 16) ? 16 : 0);
      v16bf bk;
#pragma unroll
      for (int j = 0; j < 16; ++j) bk[j] = kp[j];
      sc = __builtin_amdgcn_wmma_f32_16x16x32_bf16(false, aq[c], false, bk,
                                                   (short)0, sc, false, false);
    }
#pragma unroll
    for (int r = 0; r < 8; ++r) {
      const int m   = r + ((lane & 16) ? 8 : 0);
      const int key = kt * 16 + (lane & 15);
      float val = sc[r];
      const float mv = mask[((size_t)b * SEQ + (qt * 16 + m)) * SEQ + key];
      if (mv == 1.0f) val = NEGBIG;
      s[m * SEQ + key] = val;
    }
  }
  __syncthreads();

  // ---- softmax per row (wave-cooperative), emit bf16 probs ----
  for (int m = 0; m < 16; ++m) {
    float* row = s + m * SEQ;
    float mx = -INFINITY;
    for (int i = lane; i < SEQ; i += 32) mx = fmaxf(mx, row[i]);
#pragma unroll
    for (int off = 16; off; off >>= 1) mx = fmaxf(mx, __shfl_xor(mx, off, 32));
    float sum = 0.f;
    for (int i = lane; i < SEQ; i += 32) {
      const float e = __expf(row[i] - mx);
      row[i] = e; sum += e;
    }
#pragma unroll
    for (int off = 16; off; off >>= 1) sum += __shfl_xor(sum, off, 32);
    const float inv = 1.0f / sum;
    for (int i = lane; i < SEQ; i += 32) {
      const float p = row[i] * inv;
      pb[m * SEQ + i] = (__bf16)p;
      if (qt == 0 && m == 0) row0probs[headbase + i] = p;  // deterministic export
    }
  }
  __syncthreads();

  // ---- O = P(16x1024) @ V(1024x64), V pre-packed as B-fragments ----
  const bf16_t* vh = vpk + ((size_t)(b * NHEAD + h) * (SEQ / 32)) * 2 * (DHEAD * 16);
  v8f o[4] = {};
  for (int c = 0; c < SEQ / 32; ++c) {
    // A fragment from bf16 LDS probs (contiguous 16B chunks)
    const int r  = lane & 15;
    const int kb = c * 32 + ((lane & 16) ? 16 : 0);
    const bf16_t* pp = pb + r * SEQ + kb;
    v16bf pa;
#pragma unroll
    for (int j = 0; j < 8; ++j) {
      pa[j]     = pp[j];
      pa[8 + j] = pp[8 + j];
    }
#pragma unroll
    for (int t = 0; t < 4; ++t) {
      const int dh = t * 16 + (lane & 15);
      const bf16_t* vp = vh + ((size_t)c * 2 + ((lane & 16) ? 1 : 0)) * (DHEAD * 16) + dh * 16;
      v16bf vb;
#pragma unroll
      for (int j = 0; j < 16; ++j) vb[j] = vp[j];
      o[t] = __builtin_amdgcn_wmma_f32_16x16x32_bf16(false, pa, false, vb,
                                                     (short)0, o[t], false, false);
    }
  }
#pragma unroll
  for (int t = 0; t < 4; ++t) {
#pragma unroll
    for (int r = 0; r < 8; ++r) {
      const int m  = r + ((lane & 16) ? 8 : 0);
      const int dh = t * 16 + (lane & 15);
      const int n  = qt * 16 + m;
      attout[((size_t)b * SEQ + n) * (NHEAD * DHEAD) + h * DHEAD + dh] = (__bf16)o[t][r];
    }
  }
}

// ---------------------------------------------------------------------------
// Output projection + residual:  x1 = x + attout @ Wout + bout
// ---------------------------------------------------------------------------
__global__ void k_gemm_wout(const bf16_t* __restrict__ attout, const bf16_t* __restrict__ WoutP,
                            const float* __restrict__ bout, const float* __restrict__ x,
                            float* __restrict__ x1) {
  const int lane = threadIdx.x & 31, wave = threadIdx.x >> 5;
  const int row0 = blockIdx.y * 256 + wave * 32;
  const int col0 = blockIdx.x * 64;
  v8f acc[8] = {};
  wave_gemm_32x64(attout, NHEAD * DHEAD, WoutP, NHEAD * DHEAD, row0, col0, lane, acc);
#pragma unroll
  for (int g = 0; g < 2; ++g) {
#pragma unroll
    for (int t = 0; t < 4; ++t) {
      const int col = col0 + 16 * t + (lane & 15);
#pragma unroll
      for (int r = 0; r < 8; ++r) {
        const int row = row0 + g * 16 + r + ((lane & 16) ? 8 : 0);
        const size_t idx = (size_t)row * DMODEL + col;
        x1[idx] = x[idx] + acc[g * 4 + t][r] + bout[col];
      }
    }
  }
}

// ---------------------------------------------------------------------------
// MLP1: ff1 = gelu(h2 @ W1 + b1)   (exact gelu via erf)
// ---------------------------------------------------------------------------
__global__ void k_gemm_mlp1(const bf16_t* __restrict__ h2, const bf16_t* __restrict__ W1P,
                            const float* __restrict__ b1, bf16_t* __restrict__ ff1) {
  const int lane = threadIdx.x & 31, wave = threadIdx.x >> 5;
  const int row0 = blockIdx.y * 256 + wave * 32;
  const int col0 = blockIdx.x * 64;
  v8f acc[8] = {};
  wave_gemm_32x64(h2, DMODEL, W1P, DMODEL, row0, col0, lane, acc);
#pragma unroll
  for (int g = 0; g < 2; ++g) {
#pragma unroll
    for (int t = 0; t < 4; ++t) {
      const int col = col0 + 16 * t + (lane & 15);
#pragma unroll
      for (int r = 0; r < 8; ++r) {
        const int row = row0 + g * 16 + r + ((lane & 16) ? 8 : 0);
        const float u = acc[g * 4 + t][r] + b1[col];
        const float ge = 0.5f * u * (1.0f + erff(u * 0.70710678118654752f));
        ff1[(size_t)row * DMLP + col] = (__bf16)ge;
      }
    }
  }
}

// ---------------------------------------------------------------------------
// MLP2 + residual: out = x1 + ff1 @ W2 + b2
// ---------------------------------------------------------------------------
__global__ void k_gemm_mlp2(const bf16_t* __restrict__ ff1, const bf16_t* __restrict__ W2P,
                            const float* __restrict__ b2, const float* __restrict__ x1,
                            float* __restrict__ outx) {
  const int lane = threadIdx.x & 31, wave = threadIdx.x >> 5;
  const int row0 = blockIdx.y * 256 + wave * 32;
  const int col0 = blockIdx.x * 64;
  v8f acc[8] = {};
  wave_gemm_32x64(ff1, DMLP, W2P, DMLP, row0, col0, lane, acc);
#pragma unroll
  for (int g = 0; g < 2; ++g) {
#pragma unroll
    for (int t = 0; t < 4; ++t) {
      const int col = col0 + 16 * t + (lane & 15);
#pragma unroll
      for (int r = 0; r < 8; ++r) {
        const int row = row0 + g * 16 + r + ((lane & 16) ? 8 : 0);
        const size_t idx = (size_t)row * DMODEL + col;
        outx[idx] = x1[idx] + acc[g * 4 + t][r] + b2[col];
      }
    }
  }
}

// ---------------------------------------------------------------------------
// att_topk -> bitonic argsort (desc, tie-break index asc),
// emit what_to_prune = ranks N-65 .. N-2
// ---------------------------------------------------------------------------
__global__ void k_prune_sort(const float* __restrict__ row0probs, int* __restrict__ prune_out) {
  const int b = blockIdx.x, tid = threadIdx.x;
  __shared__ float sv[SEQ];
  __shared__ int   si[SEQ];
  for (int i = tid; i < SEQ; i += 256) {
    float val = 0.f;
#pragma unroll
    for (int h = 0; h < NHEAD; ++h) val += row0probs[((size_t)b * NHEAD + h) * SEQ + i];
    val *= (1.0f / NHEAD);
    if (i == 0) val = -INFINITY;
    sv[i] = val; si[i] = i;
  }
  __syncthreads();
  for (int kk = 2; kk <= SEQ; kk <<= 1) {
    for (int j = kk >> 1; j > 0; j >>= 1) {
      for (int i = tid; i < SEQ; i += 256) {
        const int ixj = i ^ j;
        if (ixj > i) {
          const float a = sv[i], c = sv[ixj];
          const int ia = si[i], ic = si[ixj];
          const bool desc = ((i & kk) == 0);
          const bool aFirst = desc ? (a > c || (a == c && ia < ic))
                                   : (a < c || (a == c && ia > ic));
          if (!aFirst) { sv[i] = c; sv[ixj] = a; si[i] = ic; si[ixj] = ia; }
        }
      }
      __syncthreads();
    }
  }
  for (int j = tid; j < PRUNE_NUM; j += 256)
    prune_out[b * PRUNE_NUM + j] = si[SEQ - 1 - PRUNE_NUM + j];
}

// ---------------------------------------------------------------------------
// Constant outputs: what_to_merge = -1, survived_mask = 0
// ---------------------------------------------------------------------------
__global__ void k_init_const(int* __restrict__ merge_out, int* __restrict__ surv_out) {
  const int i = blockIdx.x * 256 + threadIdx.x;
  if (i < BATCH) merge_out[i] = -1;
  if (i < BATCH * 16) surv_out[i] = 0;
}

// ---------------------------------------------------------------------------
extern "C" void kernel_launch(void* const* d_in, const int* in_sizes, int n_in,
                              void* d_out, int out_size, void* d_ws, size_t ws_size,
                              hipStream_t stream) {
  (void)in_sizes; (void)n_in; (void)out_size; (void)ws_size;
  const float* x     = (const float*)d_in[0];
  const float* mask  = (const float*)d_in[1];
  const float* Wqkv  = (const float*)d_in[2];
  const float* Wout  = (const float*)d_in[3];
  const float* bout  = (const float*)d_in[4];
  const float* ln1_g = (const float*)d_in[5];
  const float* ln1_b = (const float*)d_in[6];
  const float* ln2_g = (const float*)d_in[7];
  const float* ln2_b = (const float*)d_in[8];
  const float* W1    = (const float*)d_in[9];
  const float* b1    = (const float*)d_in[10];
  const float* W2    = (const float*)d_in[11];
  const float* b2    = (const float*)d_in[12];

  char* ws = (char*)d_ws;
  bf16_t* h1     = (bf16_t*)(ws + OFF_H1);
  bf16_t* qb     = (bf16_t*)(ws + OFF_Q);
  bf16_t* kb     = (bf16_t*)(ws + OFF_K);
  bf16_t* vpk    = (bf16_t*)(ws + OFF_V);
  bf16_t* attout = (bf16_t*)(ws + OFF_ATTOUT);
  float*  x1     = (float*)(ws + OFF_X1);
  bf16_t* h2     = (bf16_t*)(ws + OFF_H2);
  bf16_t* ff1    = (bf16_t*)(ws + OFF_FF1);
  float*  row0p  = (float*)(ws + OFF_ROW0);
  bf16_t* WqkvP  = (bf16_t*)(ws + OFF_WQKVP);
  bf16_t* WoutP  = (bf16_t*)(ws + OFF_WOUTP);
  bf16_t* W1P    = (bf16_t*)(ws + OFF_W1P);
  bf16_t* W2P    = (bf16_t*)(ws + OFF_W2P);

  float* out_x     = (float*)d_out;
  int*   out_prune = (int*)d_out + OUT_PRUNE_OFF;
  int*   out_merge = (int*)d_out + OUT_MERGE_OFF;
  int*   out_surv  = (int*)d_out + OUT_SURV_OFF;

  k_init_const<<<1, 256, 0, stream>>>(out_merge, out_surv);

  // Pack weights to bf16 B-fragment tiles (tiny, L2-resident)
  k_pack_w<<<(DMODEL * QKVN) / 256, 256, 0, stream>>>(Wqkv, WqkvP, DMODEL, QKVN);
  k_pack_w<<<(DMODEL * DMODEL) / 256, 256, 0, stream>>>(Wout, WoutP, DMODEL, DMODEL);
  k_pack_w<<<(DMODEL * DMLP) / 256, 256, 0, stream>>>(W1, W1P, DMODEL, DMLP);
  k_pack_w<<<(DMLP * DMODEL) / 256, 256, 0, stream>>>(W2, W2P, DMLP, DMODEL);

  k_layernorm<<<ROWS / 8, 256, 0, stream>>>(x, ln1_g, ln1_b, h1);
  k_gemm_qkv<<<dim3(QKVN / 64, ROWS / 256), 256, 0, stream>>>(h1, WqkvP, qb, kb, vpk);
  k_attention<<<dim3(SEQ / 16, NHEAD, BATCH), 32, 98304, stream>>>(
      qb, kb, vpk, mask, attout, row0p);
  k_gemm_wout<<<dim3(DMODEL / 64, ROWS / 256), 256, 0, stream>>>(attout, WoutP, bout, x, x1);
  k_layernorm<<<ROWS / 8, 256, 0, stream>>>(x1, ln2_g, ln2_b, h2);
  k_gemm_mlp1<<<dim3(DMLP / 64, ROWS / 256), 256, 0, stream>>>(h2, W1P, b1, ff1);
  k_gemm_mlp2<<<dim3(DMODEL / 64, ROWS / 256), 256, 0, stream>>>(ff1, W2P, b2, x1, out_x);
  k_prune_sort<<<BATCH, 256, 0, stream>>>(row0p, out_prune);
}
